// NeuromorphicLiquidNetwork_11252814315544
// MI455X (gfx1250) — matrix operand order repaired and checked by reading, not compiled, not run on hardware
//
#include <hip/hip_runtime.h>
#include <hip/hip_bf16.h>
#include <math.h>

// ---------------------------------------------------------------------------
// NeuromorphicLiquidNetwork fused pipeline for MI455X (gfx1250, wave32, WMMA)
//
// All five GEMMs run on v_wmma_f32_16x16x32_bf16 (bf16 in, f32 accumulate);
// fp32->bf16 conversion happens in-register while staging operands, so each
// weight matrix streams from HBM exactly once per launch. Memristive
// conductance transforms and |h|>0.5 activity masks are compile-time modes
// folded into the operand path of a dual-pass GEMM.
//
// Scheduling: LDS slab is double-buffered (global loads of slab k+1 issue
// before the WMMA chain of slab k; one barrier per K-step), and B fragments
// are software-pipelined (fragment nf+1's ds_load_b128 pair in flight while
// WMMA nf executes) so waits are partial instead of dscnt==0 stalls.
// ---------------------------------------------------------------------------

typedef __attribute__((ext_vector_type(16))) __bf16 v16bf;
typedef __attribute__((ext_vector_type(8)))  __bf16 v8bf;
typedef __attribute__((ext_vector_type(8)))  float  v8f;

#define KSTEP      32
#define MTILE      64                    // 4 waves x 16 rows
#define NTILE      128                   // 8 fragments of 16 cols per wave
#define LDS_STRIDE 40                    // KSTEP + 8 bf16 pad (bank spread)
#define BUFSZ      (NTILE * LDS_STRIDE)  // one slab buffer, bf16 elements

// A modes: 0 identity, 1 keep where |v|>0.5 else 0, 2 complement
template<int M> __device__ __forceinline__ float amode(float v) {
    if constexpr (M == 1) return fabsf(v) > 0.5f ? v : 0.0f;
    else if constexpr (M == 2) return fabsf(v) > 0.5f ? 0.0f : v;
    else return v;
}
// B modes: 0 identity, 1 potentiate min(v*1.05,1), 2 decay v*0.995
template<int M> __device__ __forceinline__ float bmode(float v) {
    if constexpr (M == 1) return fminf(v * 1.05f, 1.0f);
    else if constexpr (M == 2) return v * 0.995f;
    else return v;
}

// Convert a thread's 2(K) x 16(N) staged registers and scatter into the
// transposed LDS slab (two bf16 of adjacent K packed per ds_store_b32).
template<int BM>
__device__ __forceinline__ void store_slab(__bf16* dst, int nc, int kr,
                                           const float4 (&r0)[4],
                                           const float4 (&r1)[4]) {
    const float* f0 = (const float*)r0;
    const float* f1 = (const float*)r1;
#pragma unroll
    for (int j = 0; j < 16; ++j) {
        union { __bf16 h[2]; unsigned u; } pk;
        pk.h[0] = (__bf16)bmode<BM>(f0[j]);
        pk.h[1] = (__bf16)bmode<BM>(f1[j]);
        *(unsigned*)&dst[(nc + j) * LDS_STRIDE + kr] = pk.u;
    }
}

// A fragment per the 16-bit A 16x32 layout (ISA 7.12.2):
// lane half 0: K = +0..7 and +16..23 ; half 1: +8 each.
template<int AM>
__device__ __forceinline__ v16bf load_a(const float* __restrict__ arow_k, int half) {
    const float4* ap1 = (const float4*)(arow_k + half * 8);
    const float4* ap2 = (const float4*)(arow_k + 16 + half * 8);
    float4 a0 = ap1[0], a1 = ap1[1], a2 = ap2[0], a3 = ap2[1];
    const float g1[8] = {a0.x, a0.y, a0.z, a0.w, a1.x, a1.y, a1.z, a1.w};
    const float g2[8] = {a2.x, a2.y, a2.z, a2.w, a3.x, a3.y, a3.z, a3.w};
    v16bf a;
#pragma unroll
    for (int j = 0; j < 8; ++j) {
        a[j]     = (__bf16)amode<AM>(g1[j]);
        a[8 + j] = (__bf16)amode<AM>(g2[j]);
    }
    return a;
}

// One K-sweep accumulating into c[8]; AM/BM are compile-time operand modes.
template<int AM, int BM>
__device__ __forceinline__ void gemm_pass(
    const float* __restrict__ A, int lda,
    const float* __restrict__ B, int ldb, int K,
    int m0, int n0, __bf16* bT, v8f (&c)[8])
{
    const int tid  = threadIdx.x;
    const int lane = tid & 31;
    const int half = lane >> 4;
    const int l16  = lane & 15;
    // staging: each thread owns a 2(K) x 16(N) sub-tile of the 32x128 slab
    const int kr = (tid >> 3) * 2;      // even k-row: 0,2,..,30
    const int nc = (tid & 7) * 16;      // 0,16,..,112

    const float* arow  = A + (size_t)(m0 + l16) * lda;
    const float* bbase = B + (size_t)kr * ldb + n0 + nc;

    // ---- prologue: stage slab 0 into buffer 0 ------------------------------
    {
        const float4* s0 = (const float4*)(bbase);
        const float4* s1 = (const float4*)(bbase + ldb);
        float4 r0[4], r1[4];
#pragma unroll
        for (int q = 0; q < 4; ++q) { r0[q] = s0[q]; r1[q] = s1[q]; }
        store_slab<BM>(bT, nc, kr, r0, r1);
    }
    __syncthreads();

    int buf = 0;
    for (int k0 = 0; k0 < K; k0 += KSTEP) {
        const bool has_next = (k0 + KSTEP) < K;

        // ---- issue next slab's global loads first (hidden under WMMAs) ----
        float4 r0[4], r1[4];
        if (has_next) {
            const float4* s0 = (const float4*)(bbase + (size_t)(k0 + KSTEP) * ldb);
            const float4* s1 = (const float4*)(bbase + (size_t)(k0 + KSTEP + 1) * ldb);
#pragma unroll
            for (int q = 0; q < 4; ++q) { r0[q] = s0[q]; r1[q] = s1[q]; }
            if (k0 + 2 * KSTEP < K)   // global_prefetch_b8 two slabs ahead
                __builtin_prefetch((const float*)s0 + (size_t)KSTEP * ldb, 0, 1);
        }

        // ---- A fragment ---------------------------------------------------
        v16bf a = load_a<AM>(arow + k0, half);

        // ---- 8 WMMAs, B fragments software-pipelined from LDS -------------
        const __bf16* fb = bT + buf * BUFSZ + l16 * LDS_STRIDE + half * 8;
        v8bf lo = *(const v8bf*)(fb);          // frag 0, ds_load_b128 x2
        v8bf hi = *(const v8bf*)(fb + 16);
#pragma unroll
        for (int nf = 0; nf < 8; ++nf) {
            v8bf nlo = lo, nhi = hi;
            if (nf < 7) {                       // frag nf+1 in flight under WMMA nf
                const __bf16* nb = fb + (nf + 1) * 16 * LDS_STRIDE;
                nlo = *(const v8bf*)(nb);
                nhi = *(const v8bf*)(nb + 16);
            }
            v16bf b = __builtin_shufflevector(lo, hi,
                      0, 1, 2, 3, 4, 5, 6, 7, 8, 9, 10, 11, 12, 13, 14, 15);
            c[nf] = __builtin_amdgcn_wmma_f32_16x16x32_bf16(
                        false, a, false, b, (short)0, c[nf], false, false);
            lo = nlo; hi = nhi;
        }

        // ---- stage next slab into the alternate buffer --------------------
        if (has_next)
            store_slab<BM>(bT + (buf ^ 1) * BUFSZ, nc, kr, r0, r1);
        __syncthreads();
        buf ^= 1;
    }
}

template<int AM, int BM, bool HAS_BIAS, bool DUAL, int AM2, int BM2>
__global__ __launch_bounds__(128)
void nln_gemm(const float* __restrict__ A, int lda,
              const float* __restrict__ B, int ldb,
              const float* __restrict__ bias,
              float* __restrict__ C, int ldc, int K)
{
    __shared__ __bf16 bT[2 * BUFSZ];    // double-buffered 32x128 bf16 slab
    const int wave = threadIdx.x >> 5;
    const int lane = threadIdx.x & 31;
    const int half = lane >> 4;
    const int l16  = lane & 15;
    const int m0 = blockIdx.y * MTILE + wave * 16;
    const int n0 = blockIdx.x * NTILE;

    v8f c[8] = {};
    gemm_pass<AM, BM>(A, lda, B, ldb, K, m0, n0, bT, c);
    if constexpr (DUAL)
        gemm_pass<AM2, BM2>(A, lda, B, ldb, K, m0, n0, bT, c);

    // epilogue: bias + store (C layout: VGPR i -> row half*8+i, col = lane%16)
#pragma unroll
    for (int nf = 0; nf < 8; ++nf) {
        const int ncol = n0 + nf * 16 + l16;
        float bv = 0.0f;
        if constexpr (HAS_BIAS) bv = bias[ncol];
#pragma unroll
        for (int i = 0; i < 8; ++i)
            C[(size_t)(m0 + half * 8 + i) * ldc + ncol] = c[nf][i] + bv;
    }
}

// --------------------------- GRU gate elementwise ---------------------------
__global__ void nln_gru(const float* __restrict__ gi, const float* __restrict__ gh,
                        const float* __restrict__ bh_n, const float* __restrict__ ls,
                        float* __restrict__ h_new, int Bn, int Hn)
{
    int idx = blockIdx.x * blockDim.x + threadIdx.x;
    if (idx >= Bn * Hn) return;
    int b = idx / Hn, h = idx - b * Hn;
    size_t base = (size_t)b * 3 * Hn;
    float gir = gi[base + h], giz = gi[base + Hn + h], gin = gi[base + 2 * Hn + h];
    float ghr = gh[base + h], ghz = gh[base + Hn + h], ghn = gh[base + 2 * Hn + h];
    float r = 1.0f / (1.0f + __expf(-(gir + ghr)));
    float z = 1.0f / (1.0f + __expf(-(giz + ghz)));
    float n = tanhf(gin + r * (ghn + bh_n[h]));
    h_new[idx] = (1.0f - z) * n + z * ls[idx];
}

// --------------------------- LIF neuron elementwise -------------------------
__global__ void nln_lif(const float* __restrict__ membrane, const int* __restrict__ refractory,
                        const float* __restrict__ h_new, const float* __restrict__ synaptic,
                        float* __restrict__ coupled, float* __restrict__ reset_mem,
                        int* __restrict__ new_refrac, float* __restrict__ spikes, int total)
{
    int i = blockIdx.x * blockDim.x + threadIdx.x;
    if (i >= total) return;
    const float decay = 0.951229424500714f;      // exp(-1/20)
    const float one_m = 0.048770575499286f;      // 1 - decay
    float nm  = membrane[i] * decay + synaptic[i] * one_m;
    int   ref = refractory[i];
    bool  fire = (nm > 1.0f) && (ref <= 0);
    float s = fire ? 1.0f : 0.0f;
    reset_mem[i]  = fire ? 0.0f : nm;
    new_refrac[i] = fire ? 3 : (ref > 0 ? ref - 1 : 0);
    coupled[i]    = h_new[i] + s * 0.3f;
    spikes[i]     = s;
}

// ---------------------------------------------------------------------------
extern "C" void kernel_launch(void* const* d_in, const int* in_sizes, int n_in,
                              void* d_out, int out_size, void* d_ws, size_t ws_size,
                              hipStream_t stream) {
    const float* x      = (const float*)d_in[0];   // (512,1024)
    const float* ls     = (const float*)d_in[1];   // (512,2048)
    const float* membr  = (const float*)d_in[2];   // (512,2048)
    const int*   refrac = (const int*)  d_in[3];   // (512,2048) int32
    const float* W_in   = (const float*)d_in[4];   // (1024,2048)
    const float* b_in   = (const float*)d_in[5];   // (2048)
    const float* Wi     = (const float*)d_in[6];   // (2048,6144)
    const float* bi     = (const float*)d_in[7];   // (6144)
    const float* Wh     = (const float*)d_in[8];   // (2048,6144)
    const float* bh_n   = (const float*)d_in[9];   // (2048)
    const float* G      = (const float*)d_in[10];  // (2048,2048)
    const float* W_out  = (const float*)d_in[11];  // (2048,256)
    const float* b_out  = (const float*)d_in[12];  // (256)

    const int Bn = 512, DIN = 1024, Hn = 2048, DOUT = 256;

    // scratch layout (floats): u | gi | gh | h_new | syn  (~38 MB)
    float* u     = (float*)d_ws;
    float* gi    = u     + (size_t)Bn * Hn;
    float* gh    = gi    + (size_t)Bn * 3 * Hn;
    float* h_new = gh    + (size_t)Bn * 3 * Hn;
    float* syn   = h_new + (size_t)Bn * Hn;

    // outputs, concatenated flat in return order
    float* out        = (float*)d_out;                   // 512*256
    float* coupled    = out        + (size_t)Bn * DOUT;  // 512*2048
    float* reset_mem  = coupled    + (size_t)Bn * Hn;    // 512*2048
    int*   new_refrac = (int*)(reset_mem + (size_t)Bn * Hn);
    float* spikes     = reset_mem  + 2 * (size_t)Bn * Hn;

    dim3 blk(128);
    const int elems = Bn * Hn;
    const dim3 gridH (Hn     / NTILE, Bn / MTILE);   // N = 2048
    const dim3 grid3H(3 * Hn / NTILE, Bn / MTILE);   // N = 6144
    const dim3 gridO (DOUT   / NTILE, Bn / MTILE);   // N = 256

    // u = x @ W_in + b_in
    nln_gemm<0, 0, true, false, 0, 0><<<gridH, blk, 0, stream>>>(
        x, DIN, W_in, Hn, b_in, u, Hn, DIN);

    // gi = u @ Wi + bi
    nln_gemm<0, 0, true, false, 0, 0><<<grid3H, blk, 0, stream>>>(
        u, Hn, Wi, 3 * Hn, bi, gi, 3 * Hn, Hn);

    // gh = liquid_state @ Wh
    nln_gemm<0, 0, false, false, 0, 0><<<grid3H, blk, 0, stream>>>(
        ls, Hn, Wh, 3 * Hn, nullptr, gh, 3 * Hn, Hn);

    // GRU gates -> h_new
    nln_gru<<<(elems + 255) / 256, 256, 0, stream>>>(gi, gh, bh_n, ls, h_new, Bn, Hn);

    // synaptic = h_act @ min(1.05G,1) + (h_new - h_act) @ 0.995G  (dual-pass,
    // same A/B pointers, different compile-time operand transforms)
    nln_gemm<1, 1, false, true, 2, 2><<<gridH, blk, 0, stream>>>(
        h_new, Hn, G, Hn, nullptr, syn, Hn, Hn);

    // LIF update -> coupled_liquid, reset_mem, new_refrac, spikes
    nln_lif<<<(elems + 255) / 256, 256, 0, stream>>>(
        membr, refrac, h_new, syn, coupled, reset_mem, new_refrac, spikes, elems);

    // out = spikes @ W_out + b_out
    nln_gemm<0, 0, true, false, 0, 0><<<gridO, blk, 0, stream>>>(
        spikes, Hn, W_out, DOUT, b_out, out, DOUT, Hn);
}
